// GRUForward_78658031059230
// MI455X (gfx1250) — matrix operand order repaired and checked by reading, not compiled, not run on hardware
//
#include <hip/hip_runtime.h>
#include <cstdint>
#include <cstddef>

// ---------------------------------------------------------------------------
// Fused LOCF-impute + GRU scan + BatchNorm(eval) + FC head for gfx1250.
//   B=256, S=2048, I=64, H=128, O=1.
// 16 workgroups x 16 batch rows; 24 waves/WG = one 16-wide gate N-tile each.
// Weights resident in VGPRs as WMMA B-fragments; h state ping-pongs in LDS.
// Per step (2 barriers, the minimum):
//   phase B: 6 x v_wmma_f32_16x16x32_f16 per wave (3 independent 2-chains)
//   barrier
//   waves 0-7 : v_tanh/v_exp activations + h update  (serial-path math)
//   waves 8-23: GLOBAL_LOAD_ASYNC_TO_LDS_B64 t+2, LOCF-impute t+1 (overlap)
//   barrier
// ---------------------------------------------------------------------------

typedef _Float16 v16h __attribute__((ext_vector_type(16)));
typedef float    v8f  __attribute__((ext_vector_type(8)));
typedef int      v2i  __attribute__((ext_vector_type(2)));

#define S_LEN    2048
#define B_ALL    256
#define I_DIM    64
#define H_DIM    128
#define ROWS     16
#define WAVES    24
#define NTHREADS (WAVES * 32)

#if defined(__HIP_DEVICE_COMPILE__) && \
    __has_builtin(__builtin_amdgcn_global_load_async_to_lds_b64) && \
    __has_builtin(__builtin_amdgcn_s_wait_asynccnt)
#define USE_ASYNC_LDS 1
#define GLOBAL_AS __attribute__((address_space(1)))
#define LDS_AS    __attribute__((address_space(3)))
#else
#define USE_ASYNC_LDS 0
#endif

union AFrag { v16h v; _Float16 h[16]; uint4 q[2]; };
union BFrag { v16h v; _Float16 h[16]; };

struct __align__(16) SharedState {
  float     last[ROWS][I_DIM];        // LOCF carry state (f32)            4 KB
  _Float16  xt[ROWS][I_DIM];          // imputed x_t tile (f16, row-major) 2 KB
  float     hbuf[ROWS][H_DIM];        // h state f32                       8 KB
  _Float16  hh[ROWS][H_DIM];          // h state f16 (WMMA A source)       4 KB
  float     gbuf[4][ROWS][H_DIM];     // r, z, n_x, n_h pre-activations   32 KB
  float     xstage[2][ROWS][I_DIM];   // async x staging (double buffer)   8 KB
  int       mstage[2][ROWS][I_DIM];   // async mask staging                8 KB
  float     br[H_DIM], bz[H_DIM];     // fused biases                      1 KB
  float     bnx[H_DIM], bnh[H_DIM];   //                                   1 KB
  float     outacc[ROWS];             // FC-head accumulators
};

__device__ __forceinline__ float sigf(float v) {
#if defined(__HIP_DEVICE_COMPILE__) && __has_builtin(__builtin_amdgcn_rcpf)
  return __builtin_amdgcn_rcpf(1.0f + __expf(-v));   // v_rcp_f32 + v_exp_f32
#else
  return 1.0f / (1.0f + __expf(-v));
#endif
}

__device__ __forceinline__ float fast_tanh(float v) {
#if defined(__HIP_DEVICE_COMPILE__) && __has_builtin(__builtin_amdgcn_tanhf)
  return __builtin_amdgcn_tanhf(v);                  // gfx1250 v_tanh_f32
#elif defined(__HIP_DEVICE_COMPILE__) && __has_builtin(__builtin_amdgcn_tanh_f32)
  return __builtin_amdgcn_tanh_f32(v);
#else
  return tanhf(v);
#endif
}

__global__ __launch_bounds__(NTHREADS, 1)
void gru_locf_wmma_kernel(const float* __restrict__ x,     const int* __restrict__ mask,
                          const float* __restrict__ x_mean,
                          const float* __restrict__ w_ih,  const float* __restrict__ w_hh,
                          const float* __restrict__ b_ih,  const float* __restrict__ b_hh,
                          const float* __restrict__ bn_g,  const float* __restrict__ bn_b,
                          const float* __restrict__ bn_m,  const float* __restrict__ bn_v,
                          const float* __restrict__ fc_w,  const float* __restrict__ fc_b,
                          float* __restrict__ out)
{
  __shared__ SharedState sm;

  const int tid  = threadIdx.x;
  const int wave = tid >> 5;
  const int lane = tid & 31;
  const int b0   = blockIdx.x * ROWS;

  // ---- init shared state ----------------------------------------------------
  for (int idx = tid; idx < ROWS * I_DIM; idx += NTHREADS)
    sm.last[idx / I_DIM][idx % I_DIM] = x_mean[idx % I_DIM];
  for (int idx = tid; idx < ROWS * H_DIM; idx += NTHREADS) {
    const int r = idx / H_DIM, c = idx % H_DIM;
    sm.hbuf[r][c] = 0.0f;
    sm.hh[r][c]   = (_Float16)0.0f;
  }
  for (int h = tid; h < H_DIM; h += NTHREADS) {
    sm.br[h]  = b_ih[h]            + b_hh[h];             // r gate: bias fused
    sm.bz[h]  = b_ih[H_DIM + h]    + b_hh[H_DIM + h];     // z gate: bias fused
    sm.bnx[h] = b_ih[2 * H_DIM + h];                      // n gate x-part bias
    sm.bnh[h] = b_hh[2 * H_DIM + h];                      // n gate h-part bias (scaled by r)
  }

  // ---- load this wave's weight B-fragments into VGPRs (resident all steps) --
  // B-fragment k order used here (contiguous, same as A): lanes 0-15 hold K=0..15 (col N=lane),
  // lanes 16-31 hold K=16..31 (col N=lane-16); 2 K values per VGPR.
  const int g = wave >> 3;                      // 0=r, 1=z, 2=n
  const int s = wave & 7;                       // H sub-tile
  const int nrow  = g * H_DIM + s * 16 + (lane & 15);   // weight-matrix row (= N col)
  const int khalf = (lane >> 4) * 16;                   // this lane's K base
  BFrag bw[6];                                  // kt 0-1: w_ih (K=64), kt 2-5: w_hh (K=128)
  #pragma unroll
  for (int kt = 0; kt < 2; ++kt) {
    const float* src = w_ih + (size_t)nrow * I_DIM + kt * 32 + khalf;
    #pragma unroll
    for (int j = 0; j < 16; ++j) bw[kt].h[j] = (_Float16)src[j];
  }
  #pragma unroll
  for (int kt = 0; kt < 4; ++kt) {
    const float* src = w_hh + (size_t)nrow * H_DIM + kt * 32 + khalf;
    #pragma unroll
    for (int j = 0; j < 16; ++j) bw[2 + kt].h[j] = (_Float16)src[j];
  }

  // ---- imputation-lane mapping for waves 8-23 (512 threads, 2 elems each) ---
  const int u  = (tid >= 256) ? (tid - 256) : 0;   // 0..511
  const int wr = u >> 5;                           // batch row 0..15
  const int wi = (u << 1) & 63;                    // even column in [0,64)

#if USE_ASYNC_LDS
  // Issue GLOBAL_LOAD_ASYNC_TO_LDS_B64 for this thread's (x, mask) pair of
  // step `t` into staging buffer `buf`.  Tracked with ASYNCcnt; async loads
  // complete in order, so s_wait_asynccnt<=2 guarantees the older pair landed.
  auto issue_async = [&](int t, int buf) {
    const size_t off = ((size_t)(b0 + wr) * S_LEN + t) * I_DIM + wi;
    __builtin_amdgcn_global_load_async_to_lds_b64(
        (GLOBAL_AS v2i*)(x + off),    (LDS_AS v2i*)&sm.xstage[buf][wr][wi], 0, 0);
    __builtin_amdgcn_global_load_async_to_lds_b64(
        (GLOBAL_AS v2i*)(mask + off), (LDS_AS v2i*)&sm.mstage[buf][wr][wi], 0, 0);
  };
  auto impute_from_stage = [&](int buf) {
    float l0 = sm.last[wr][wi], l1 = sm.last[wr][wi + 1];
    const float x0 = sm.xstage[buf][wr][wi], x1 = sm.xstage[buf][wr][wi + 1];
    const int   m0 = sm.mstage[buf][wr][wi], m1 = sm.mstage[buf][wr][wi + 1];
    l0 = (m0 > 0) ? x0 : l0;
    l1 = (m1 > 0) ? x1 : l1;
    sm.last[wr][wi]     = l0;  sm.last[wr][wi + 1] = l1;
    sm.xt[wr][wi]       = (_Float16)l0;
    sm.xt[wr][wi + 1]   = (_Float16)l1;
  };
#else
  auto impute_direct = [&](int t) {
    const size_t off = ((size_t)(b0 + wr) * S_LEN + t) * I_DIM + wi;
    const float2 xv = *(const float2*)(x + off);
    const int2   mv = *(const int2*)(mask + off);
    float l0 = sm.last[wr][wi], l1 = sm.last[wr][wi + 1];
    l0 = (mv.x > 0) ? xv.x : l0;
    l1 = (mv.y > 0) ? xv.y : l1;
    sm.last[wr][wi]     = l0;  sm.last[wr][wi + 1] = l1;
    sm.xt[wr][wi]       = (_Float16)l0;
    sm.xt[wr][wi + 1]   = (_Float16)l1;
    if (t + 1 < S_LEN) {                       // global_prefetch_b8 next step
      __builtin_prefetch(x + off + I_DIM, 0, 1);
      __builtin_prefetch(mask + off + I_DIM, 0, 1);
    }
  };
#endif

  __syncthreads();

  // ---- prologue: stage + impute t=0 (waves 8-23) ----------------------------
  if (wave >= 8) {
#if USE_ASYNC_LDS
    issue_async(0, 0);
    issue_async(1, 1);
    __builtin_amdgcn_s_wait_asynccnt(2);       // t=0 pair arrived
    impute_from_stage(0);
#else
    impute_direct(0);
#endif
  }
  __syncthreads();

  // A-fragment k order used here (contiguous, same as bw): row M = lane%16;
  // lane<16 holds K=base+0..15 (q[0],q[1]); lane>=16: K=base+16..31.
  const int arow = lane & 15;
  const int akb  = (lane >> 4) * 16;

  for (int t = 0; t < S_LEN; ++t) {
    // ---- phase B: gate GEMM, 3 independent 2-WMMA chains per wave -----------
    AFrag ax0, ax1;
    ax0.q[0] = *(const uint4*)&sm.xt[arow][akb];
    ax0.q[1] = *(const uint4*)&sm.xt[arow][akb + 8];
    ax1.q[0] = *(const uint4*)&sm.xt[arow][32 + akb];
    ax1.q[1] = *(const uint4*)&sm.xt[arow][32 + akb + 8];
    AFrag ah[4];
    #pragma unroll
    for (int kt = 0; kt < 4; ++kt) {
      ah[kt].q[0] = *(const uint4*)&sm.hh[arow][kt * 32 + akb];
      ah[kt].q[1] = *(const uint4*)&sm.hh[arow][kt * 32 + akb + 8];
    }

    // cx = x-part, c0+c1 = h-part; chains are independent so the six WMMA
    // issues pipeline with only one RAW hazard per chain.
    v8f cx = {}, c0 = {}, c1 = {};
    cx = __builtin_amdgcn_wmma_f32_16x16x32_f16(false, ax0.v,   false, bw[0].v, (short)0, cx, false, false);
    c0 = __builtin_amdgcn_wmma_f32_16x16x32_f16(false, ah[0].v, false, bw[2].v, (short)0, c0, false, false);
    c1 = __builtin_amdgcn_wmma_f32_16x16x32_f16(false, ah[2].v, false, bw[4].v, (short)0, c1, false, false);
    cx = __builtin_amdgcn_wmma_f32_16x16x32_f16(false, ax1.v,   false, bw[1].v, (short)0, cx, false, false);
    c0 = __builtin_amdgcn_wmma_f32_16x16x32_f16(false, ah[1].v, false, bw[3].v, (short)0, c0, false, false);
    c1 = __builtin_amdgcn_wmma_f32_16x16x32_f16(false, ah[3].v, false, bw[5].v, (short)0, c1, false, false);

    // C-tile lane mapping: col N = lane%16, row M = v + 8*(lane/16).
    const int crow = 8 * (lane >> 4);
    const int ccol = s * 16 + (lane & 15);

    if (g < 2) {
      // r / z gates: x-part and h-part merge (bias-fused later).
      #pragma unroll
      for (int vv = 0; vv < 8; ++vv)
        sm.gbuf[g][vv + crow][ccol] = cx[vv] + c0[vv] + c1[vv];
    } else {
      // n gate: n = tanh(gx_n + r * gh_n) -> keep x/h contributions separate.
      #pragma unroll
      for (int vv = 0; vv < 8; ++vv) {
        sm.gbuf[2][vv + crow][ccol] = cx[vv];
        sm.gbuf[3][vv + crow][ccol] = c0[vv] + c1[vv];
      }
    }
    __syncthreads();

    if (wave < 8) {
      // ---- phase C: activations + h update (waves 0-7, 2048 elems) ----------
      const int col = wave * 16 + (lane & 15);
      #pragma unroll
      for (int vv = 0; vv < 8; ++vv) {
        const int row = vv + 8 * (lane >> 4);
        const float rv = sigf(sm.gbuf[0][row][col] + sm.br[col]);
        const float zv = sigf(sm.gbuf[1][row][col] + sm.bz[col]);
        const float nv = fast_tanh(sm.gbuf[2][row][col] + sm.bnx[col]
                                   + rv * (sm.gbuf[3][row][col] + sm.bnh[col]));
        const float hp = sm.hbuf[row][col];
        const float hn = (1.0f - zv) * nv + zv * hp;
        sm.hbuf[row][col] = hn;
        sm.hh[row][col]   = (_Float16)hn;
      }
    } else {
      // ---- overlapped: stage t+2, impute t+1 (waves 8-23) -------------------
#if USE_ASYNC_LDS
      if (t + 2 < S_LEN) {
        issue_async(t + 2, t & 1);               // reuse buffer consumed last window
        __builtin_amdgcn_s_wait_asynccnt(2);     // ensures t+1 pair has landed
      } else {
        __builtin_amdgcn_s_wait_asynccnt(0);
      }
      if (t + 1 < S_LEN) impute_from_stage((t + 1) & 1);
#else
      if (t + 1 < S_LEN) impute_direct(t + 1);
#endif
    }
    __syncthreads();
  }

  // ---- BatchNorm1d (eval) + FC(H -> 1) head ---------------------------------
  if (tid < ROWS) sm.outacc[tid] = 0.0f;
  __syncthreads();
  for (int idx = tid; idx < ROWS * H_DIM; idx += NTHREADS) {
    const int r = idx / H_DIM, hc = idx % H_DIM;
    const float scale = bn_g[hc] * rsqrtf(bn_v[hc] + 1e-5f);
    const float hn = (sm.hbuf[r][hc] - bn_m[hc]) * scale + bn_b[hc];
    atomicAdd(&sm.outacc[r], hn * fc_w[hc]);      // ds_add_f32
  }
  __syncthreads();
  if (tid < ROWS) out[b0 + tid] = sm.outacc[tid] + fc_b[0];
}

extern "C" void kernel_launch(void* const* d_in, const int* in_sizes, int n_in,
                              void* d_out, int out_size, void* d_ws, size_t ws_size,
                              hipStream_t stream) {
  (void)in_sizes; (void)n_in; (void)d_ws; (void)ws_size; (void)out_size;
  const float* x      = (const float*)d_in[0];
  const int*   mask   = (const int*)  d_in[1];
  /* d_in[2] = delta: accepted but unused by the reference forward */
  const float* x_mean = (const float*)d_in[3];
  const float* w_ih   = (const float*)d_in[4];
  const float* w_hh   = (const float*)d_in[5];
  const float* b_ih   = (const float*)d_in[6];
  const float* b_hh   = (const float*)d_in[7];
  const float* bn_g   = (const float*)d_in[8];
  const float* bn_b   = (const float*)d_in[9];
  const float* bn_m   = (const float*)d_in[10];
  const float* bn_v   = (const float*)d_in[11];
  const float* fc_w   = (const float*)d_in[12];
  const float* fc_b   = (const float*)d_in[13];
  float* out = (float*)d_out;

  gru_locf_wmma_kernel<<<dim3(B_ALL / ROWS), dim3(NTHREADS), 0, stream>>>(
      x, mask, x_mean, w_ih, w_hh, b_ih, b_hh,
      bn_g, bn_b, bn_m, bn_v, fc_w, fc_b, out);
}
